// MotionTransformer_14809047236894
// MI455X (gfx1250) — matrix operand order repaired and checked by reading, not compiled, not run on hardware
//
#include <hip/hip_runtime.h>
#include <hip/hip_bf16.h>
#include <math.h>

// ---------------------------------------------------------------------------
// MotionTransformer forward on gfx1250 (CDNA5, wave32).
//  - all big GEMMs: v_wmma_f32_16x16x32_bf16, fp32 accumulate
//  - GEMM tiles staged into LDS by the Tensor Data Mover (tensor_load_to_lds),
//    double-buffered with s_wait_tensorcnt 1 to overlap DMA and WMMA
//  - weights pre-converted/transposed to bf16 [N,K] once per layer
//  - WMMA fragments fed by contiguous 16B-aligned LDS reads (ds_load_b128)
// ---------------------------------------------------------------------------

typedef __bf16 bf16_t;
typedef __attribute__((ext_vector_type(16))) __bf16    v16bf;
typedef __attribute__((ext_vector_type(8)))  float     v8f;
typedef __attribute__((ext_vector_type(4)))  unsigned  v4u;
typedef __attribute__((ext_vector_type(8)))  unsigned  v8u;

#define DEV __device__ __forceinline__

constexpr int Bb   = 32;
constexpr int Tt   = 512;
constexpr int Dd   = 512;
constexpr int NTXT = 77;
constexpr int LTXT = 256;
constexpr int Hh   = 8;
constexpr int DH   = 64;
constexpr int FFNd = 2048;
constexpr int TEd  = 2048;
constexpr int NL   = 8;
constexpr int Mx   = Bb * Tt;          // 16384 rows

// ----- bf16 helpers (RNE convert) -----
DEV unsigned short f2bf_u(float f) {
  unsigned u = __builtin_bit_cast(unsigned, f);
  return (unsigned short)((u + 0x7FFFu + ((u >> 16) & 1u)) >> 16);
}
DEV bf16_t f2bf(float f) { unsigned short s = f2bf_u(f); return __builtin_bit_cast(bf16_t, s); }
DEV unsigned pack2bf(float a, float b) {
  return (unsigned)f2bf_u(a) | ((unsigned)f2bf_u(b) << 16);
}

// ----- WMMA wrapper -----
DEV v8f wmma_bf16(v16bf a, v16bf b, v8f c) {
  return __builtin_amdgcn_wmma_f32_16x16x32_bf16(
      false, a, false, b, (short)0, c, false, false);
}

// ----- Tensor Data Mover: 2D tile Global -> LDS (bf16 elements) -----
// D# group0: count=1 | lds_addr | global_addr(57b) | type=2
DEV v4u tdm_g0(unsigned lds_off, unsigned long long gaddr) {
  v4u g;
  g[0] = 1u;
  g[1] = lds_off;
  g[2] = (unsigned)gaddr;
  g[3] = (unsigned)((gaddr >> 32) & 0x1FFFFFFu) | (2u << 30);
  return g;
}
// D# group1: data_size=2B, 2D tile (tile0 x tile1), tensor dims == tile dims,
// row stride = stride0 elements. No padding / barrier / multicast / iterate.
DEV v8u tdm_g1(unsigned tile0, unsigned tile1, unsigned long long stride0) {
  v8u g;
  g[0] = 0x00010000u;                                   // data_size = 2 bytes
  g[1] = (tile0 & 0xFFFFu) << 16;                       // tensor_dim0[15:0]
  g[2] = (tile0 >> 16) | ((tile1 & 0xFFFFu) << 16);     // td0[31:16] | td1[15:0]
  g[3] = (tile1 >> 16) | (tile0 << 16);                 // td1[31:16] | tile_dim0
  g[4] = tile1;                                         // tile_dim1 (tile_dim2=0)
  g[5] = (unsigned)stride0;                             // dim0_stride[31:0]
  g[6] = (unsigned)((stride0 >> 32) & 0xFFFFu);         // dim0_stride[47:32]
  g[7] = 0u;
  return g;
}
DEV void tdm_load(v4u g0, v8u g1) {
  asm volatile("tensor_load_to_lds %0, %1" :: "s"(g0), "s"(g1) : "memory");
}
DEV unsigned lds_off_of(const void* p) {          // flat LDS addr: low 32 = offset
  return (unsigned)(uintptr_t)p;
}

// ----- vectorized fragment loads from LDS (row stride 32 bf16 = 64B) -----
DEV v16bf load_a_frag32(const bf16_t* As, int row0, int lane) {
  const bf16_t* p = As + (size_t)(row0 + (lane & 15)) * 32 + (lane >> 4) * 8;
  union { struct { v4u a, b; } u; v16bf v; } cvt;
  cvt.u.a = *(const v4u*)(p);          // K = kh .. kh+7
  cvt.u.b = *(const v4u*)(p + 16);     // K = 16+kh .. 16+kh+7
  return cvt.v;
}
DEV v16bf load_b_frag32(const bf16_t* Bt, int col0, int lane) {
  const bf16_t* p = Bt + (size_t)(col0 + (lane & 15)) * 32 + (lane >> 4) * 8;
  union { struct { v4u a, b; } u; v16bf v; } cvt;
  cvt.u.a = *(const v4u*)(p);
  cvt.u.b = *(const v4u*)(p + 16);
  return cvt.v;
}

DEV void store_out(float* p, float v)  { *p = v; }
DEV void store_out(bf16_t* p, float v) { *p = f2bf(v); }

// ---------------------------------------------------------------------------
// TDM-fed GEMM: C[M,N] = epi( A_bf16[M,K] @ Wt_bf16[N,K]^T + bias[N] )
// EPI: 0=none, 1=exact GELU, 2=residual add. N%64==0, K%32==0, M guarded at
// store (TDM may read a few rows past M into workspace slack; harmless).
// Double-buffered TDM pipeline: wave0 DMAs A tiles, wave1 DMAs B tiles;
// s_wait_tensorcnt 1 keeps one DMA in flight behind the WMMA compute.
// ---------------------------------------------------------------------------
template <int EPI, typename OutT>
__global__ __launch_bounds__(256) void gemm_tdm_k(
    const bf16_t* __restrict__ A, const bf16_t* __restrict__ Wt,
    const float* __restrict__ bias, const float* __restrict__ resid,
    OutT* __restrict__ C, int M, int N, int K) {
  __shared__ alignas(32) bf16_t As[2][128 * 32];
  __shared__ alignas(32) bf16_t Bs[2][64 * 32];
  const int tid = threadIdx.x, lane = tid & 31, wave = tid >> 5;
  const int wm = wave & 3, wn = wave >> 2;
  const int m0 = blockIdx.y * 128, n0 = blockIdx.x * 64;
  v8f acc[2][2] = {};
  const unsigned ldsA[2] = { lds_off_of(&As[0][0]), lds_off_of(&As[1][0]) };
  const unsigned ldsB[2] = { lds_off_of(&Bs[0][0]), lds_off_of(&Bs[1][0]) };
  const v8u g1A = tdm_g1(32u, 128u, (unsigned long long)K);
  const v8u g1B = tdm_g1(32u, 64u,  (unsigned long long)K);
  const unsigned long long baseA = (unsigned long long)(uintptr_t)(A + (size_t)m0 * K);
  const unsigned long long baseB = (unsigned long long)(uintptr_t)(Wt + (size_t)n0 * K);
  const int nk = K >> 5;

  // prologue: stage tile 0 into buffer 0
  if (wave == 0)      tdm_load(tdm_g0(ldsA[0], baseA), g1A);
  else if (wave == 1) tdm_load(tdm_g0(ldsB[0], baseB), g1B);

  for (int i = 0; i < nk; i++) {
    const int cur = i & 1;
    if (i + 1 < nk) {
      // stage tile i+1 into the other buffer, then wait for tile i only
      if (wave == 0)      tdm_load(tdm_g0(ldsA[cur ^ 1], baseA + 64ull * (i + 1)), g1A);
      else if (wave == 1) tdm_load(tdm_g0(ldsB[cur ^ 1], baseB + 64ull * (i + 1)), g1B);
      __builtin_amdgcn_s_wait_tensorcnt(1);
    } else {
      __builtin_amdgcn_s_wait_tensorcnt(0);
    }
    __syncthreads();                       // tile i visible to all waves

    v16bf af0 = load_a_frag32(As[cur], wm * 32,      lane);
    v16bf af1 = load_a_frag32(As[cur], wm * 32 + 16, lane);
    v16bf bf0 = load_b_frag32(Bs[cur], wn * 32,      lane);
    v16bf bf1 = load_b_frag32(Bs[cur], wn * 32 + 16, lane);
    acc[0][0] = wmma_bf16(af0, bf0, acc[0][0]);
    acc[0][1] = wmma_bf16(af0, bf1, acc[0][1]);
    acc[1][0] = wmma_bf16(af1, bf0, acc[1][0]);
    acc[1][1] = wmma_bf16(af1, bf1, acc[1][1]);
    __syncthreads();                       // done reading buf[cur] before reuse
  }

  const int mbase = m0 + wm * 32 + ((lane >> 4) << 3);
  const int nbase = n0 + wn * 32 + (lane & 15);
#pragma unroll
  for (int fi = 0; fi < 2; fi++)
#pragma unroll
    for (int fj = 0; fj < 2; fj++)
#pragma unroll
      for (int j = 0; j < 8; j++) {
        int m = mbase + fi * 16 + j;
        int n = nbase + fj * 16;
        if (m < M) {
          float v = acc[fi][fj][j] + bias[n];
          if (EPI == 1)      v = 0.5f * v * (1.0f + erff(v * 0.70710678118654752f));
          else if (EPI == 2) v += resid[(size_t)m * N + n];
          store_out(&C[(size_t)m * N + n], v);
        }
      }
}

// ---------------------------------------------------------------------------
// att[b,h,d,l] = sum_t k[b,t,h,d] * v[b,t,h,l]   (K-dim = tokens T)
// LDS fills pack 2 bf16 per dword store (ds_store_b32).
// ---------------------------------------------------------------------------
__global__ __launch_bounds__(128) void attn_kv_k(
    const float* __restrict__ Km, const float* __restrict__ Vm,
    float* __restrict__ KV, int T, int Dtot) {
  const int bh = blockIdx.x, b = bh >> 3, h = bh & 7;
  __shared__ alignas(32) bf16_t Ks[64 * 32];   // A = k^T  [d][t_local]
  __shared__ alignas(32) bf16_t Vs[64 * 32];   // B^T      [l][t_local]
  const int tid = threadIdx.x, lane = tid & 31, wave = tid >> 5;
  const int wm = wave & 1, wn = wave >> 1;
  v8f acc[2][2] = {};

  for (int t0 = 0; t0 < T; t0 += 32) {
#pragma unroll
    for (int i = 0; i < 8; i++) {             // Ks: 64 d x 16 t-pairs
      int idx = i * 128 + tid;
      int d = idx >> 4, jp = idx & 15;
      int t = t0 + 2 * jp;
      float a0 = (t     < T) ? Km[(size_t)(b * T + t    ) * Dtot + h * 64 + d] : 0.f;
      float a1 = (t + 1 < T) ? Km[(size_t)(b * T + t + 1) * Dtot + h * 64 + d] : 0.f;
      *(unsigned*)&Ks[d * 32 + 2 * jp] = pack2bf(a0, a1);
    }
#pragma unroll
    for (int i = 0; i < 8; i++) {             // Vs: 64 l x 16 t-pairs
      int idx = i * 128 + tid;
      int c = idx >> 4, rp = idx & 15;
      int t = t0 + 2 * rp;
      float a0 = (t     < T) ? Vm[(size_t)(b * T + t    ) * Dtot + h * 64 + c] : 0.f;
      float a1 = (t + 1 < T) ? Vm[(size_t)(b * T + t + 1) * Dtot + h * 64 + c] : 0.f;
      *(unsigned*)&Vs[c * 32 + 2 * rp] = pack2bf(a0, a1);
    }
    __syncthreads();
    v16bf af0 = load_a_frag32(Ks, wm * 32,      lane);
    v16bf af1 = load_a_frag32(Ks, wm * 32 + 16, lane);
    v16bf bf0 = load_b_frag32(Vs, wn * 32,      lane);
    v16bf bf1 = load_b_frag32(Vs, wn * 32 + 16, lane);
    acc[0][0] = wmma_bf16(af0, bf0, acc[0][0]);
    acc[0][1] = wmma_bf16(af0, bf1, acc[0][1]);
    acc[1][0] = wmma_bf16(af1, bf0, acc[1][0]);
    acc[1][1] = wmma_bf16(af1, bf1, acc[1][1]);
    __syncthreads();
  }
  const int mbase = wm * 32 + ((lane >> 4) << 3);
  const int nbase = wn * 32 + (lane & 15);
#pragma unroll
  for (int fi = 0; fi < 2; fi++)
#pragma unroll
    for (int fj = 0; fj < 2; fj++)
#pragma unroll
      for (int j = 0; j < 8; j++) {
        int m = mbase + fi * 16 + j;
        int n = nbase + fj * 16;
        KV[((size_t)bh * 64 + m) * 64 + n] = acc[fi][fj][j];
      }
}

// ---------------------------------------------------------------------------
// y[b,t,h,l] = sum_d q[b,t,h,d] * att[b,h,d,l]   (K = 64)
// ---------------------------------------------------------------------------
__global__ __launch_bounds__(256) void attn_qatt_k(
    const float* __restrict__ Q, const float* __restrict__ KV,
    float* __restrict__ Y, int T, int Dtot) {
  const int bh = blockIdx.x, b = bh >> 3, h = bh & 7;
  const int t0 = blockIdx.y * 128;
  __shared__ alignas(32) bf16_t Qs[128 * 32];
  __shared__ alignas(32) bf16_t Ats[64 * 32];   // att^T [l][d_local]
  const int tid = threadIdx.x, lane = tid & 31, wave = tid >> 5;
  const int wm = wave & 3, wn = wave >> 2;
  v8f acc[2][2] = {};

  for (int k0 = 0; k0 < 64; k0 += 32) {
#pragma unroll
    for (int i = 0; i < 8; i++) {             // Qs: 128 t x 16 k-pairs
      int idx = i * 256 + tid;
      int r = idx >> 4, cp = idx & 15;
      int t = t0 + r;
      float a0 = 0.f, a1 = 0.f;
      if (t < T) {
        const float* qp = &Q[(size_t)(b * T + t) * Dtot + h * 64 + k0 + 2 * cp];
        a0 = qp[0]; a1 = qp[1];
      }
      *(unsigned*)&Qs[r * 32 + 2 * cp] = pack2bf(a0, a1);
    }
#pragma unroll
    for (int i = 0; i < 4; i++) {             // Ats: 64 l x 16 d-pairs
      int idx = i * 256 + tid;
      int c = idx >> 4, rp = idx & 15;
      float a0 = KV[((size_t)bh * 64 + k0 + 2 * rp    ) * 64 + c];
      float a1 = KV[((size_t)bh * 64 + k0 + 2 * rp + 1) * 64 + c];
      *(unsigned*)&Ats[c * 32 + 2 * rp] = pack2bf(a0, a1);
    }
    __syncthreads();
    v16bf af0 = load_a_frag32(Qs, wm * 32,      lane);
    v16bf af1 = load_a_frag32(Qs, wm * 32 + 16, lane);
    v16bf bf0 = load_b_frag32(Ats, wn * 32,      lane);
    v16bf bf1 = load_b_frag32(Ats, wn * 32 + 16, lane);
    acc[0][0] = wmma_bf16(af0, bf0, acc[0][0]);
    acc[0][1] = wmma_bf16(af0, bf1, acc[0][1]);
    acc[1][0] = wmma_bf16(af1, bf0, acc[1][0]);
    acc[1][1] = wmma_bf16(af1, bf1, acc[1][1]);
    __syncthreads();
  }
  const int mbase = t0 + wm * 32 + ((lane >> 4) << 3);
  const int nbase = wn * 32 + (lane & 15);
#pragma unroll
  for (int fi = 0; fi < 2; fi++)
#pragma unroll
    for (int fj = 0; fj < 2; fj++)
#pragma unroll
      for (int j = 0; j < 8; j++) {
        int m = mbase + fi * 16 + j;
        int n = nbase + fj * 16;
        if (m < T)
          Y[(size_t)(b * T + m) * Dtot + h * 64 + n] = acc[fi][fj][j];
      }
}

// ----- block reduction helper (blockDim.x == 256) -----
DEV float block_reduce_sum(float v, float* red) {
  int tid = threadIdx.x;
  red[tid] = v; __syncthreads();
  for (int s = 128; s > 0; s >>= 1) {
    if (tid < s) red[tid] += red[tid + s];
    __syncthreads();
  }
  float r = red[0]; __syncthreads();
  return r;
}

// LayerNorm (fp32 in) -> bf16 out. One block per row.
__global__ __launch_bounds__(256) void layernorm_bf16_k(
    const float* __restrict__ X, const float* __restrict__ g,
    const float* __restrict__ bta, bf16_t* __restrict__ O, int Dn) {
  __shared__ float red[256];
  const int row = blockIdx.x;
  const float* xr = X + (size_t)row * Dn;
  float s = 0.f;
  for (int i = threadIdx.x; i < Dn; i += 256) s += xr[i];
  float mean = block_reduce_sum(s, red) / (float)Dn;
  float vv = 0.f;
  for (int i = threadIdx.x; i < Dn; i += 256) { float d = xr[i] - mean; vv += d * d; }
  float var = block_reduce_sum(vv, red) / (float)Dn;
  float inv = rsqrtf(var + 1e-5f);
  for (int i = threadIdx.x; i < Dn; i += 256)
    O[(size_t)row * Dn + i] = f2bf((xr[i] - mean) * inv * g[i] + bta[i]);
}

// StylizationBlock middle: O = bf16( silu( LN(y)*(1+scale_b) + shift_b ) )
__global__ __launch_bounds__(256) void mod_silu_k(
    const float* __restrict__ Y, const float* __restrict__ ng,
    const float* __restrict__ nb, const float* __restrict__ eSS,
    bf16_t* __restrict__ O) {
  __shared__ float red[256];
  const int row = blockIdx.x;
  const int b = row >> 9;                       // T = 512
  const float* yr = Y + (size_t)row * Dd;
  float s = 0.f;
  for (int i = threadIdx.x; i < Dd; i += 256) s += yr[i];
  float mean = block_reduce_sum(s, red) / (float)Dd;
  float vv = 0.f;
  for (int i = threadIdx.x; i < Dd; i += 256) { float d = yr[i] - mean; vv += d * d; }
  float var = block_reduce_sum(vv, red) / (float)Dd;
  float inv = rsqrtf(var + 1e-5f);
  for (int i = threadIdx.x; i < Dd; i += 256) {
    float val = (yr[i] - mean) * inv * ng[i] + nb[i];
    float sc = eSS[b * (2 * Dd) + i];
    float sh = eSS[b * (2 * Dd) + Dd + i];
    val = val * (1.f + sc) + sh;
    O[(size_t)row * Dd + i] = f2bf(val / (1.f + __expf(-val)));   // SiLU
  }
}

// softmax over last dim (dh=64), contiguous rows. In place.
__global__ __launch_bounds__(64) void softmax_dh_k(float* __restrict__ Q) {
  __shared__ float red[64];
  const int tid = threadIdx.x;
  float* row = Q + (size_t)blockIdx.x * 64;
  float v = row[tid];
  red[tid] = v; __syncthreads();
  for (int s = 32; s > 0; s >>= 1) { if (tid < s) red[tid] = fmaxf(red[tid], red[tid + s]); __syncthreads(); }
  float mx = red[0]; __syncthreads();
  float e = __expf(v - mx);
  red[tid] = e; __syncthreads();
  for (int s = 32; s > 0; s >>= 1) { if (tid < s) red[tid] += red[tid + s]; __syncthreads(); }
  float sm = red[0]; __syncthreads();
  row[tid] = e / sm;
}

// softmax over token axis (stride Dtot), optional mask penalty. In place.
__global__ __launch_bounds__(128) void softmax_tok_k(
    float* __restrict__ Km, const float* __restrict__ mask, int T, int Dtot) {
  __shared__ float red[128];
  const int tid = threadIdx.x;
  const int b = blockIdx.x / Dtot, col = blockIdx.x % Dtot;
  float* base = Km + (size_t)b * T * Dtot + col;
  float mx = -1e30f;
  for (int t = tid; t < T; t += 128) {
    float v = base[(size_t)t * Dtot];
    if (mask) v += (1.f - mask[b * T + t]) * (-1000000.0f);
    mx = fmaxf(mx, v);
  }
  red[tid] = mx; __syncthreads();
  for (int s = 64; s > 0; s >>= 1) { if (tid < s) red[tid] = fmaxf(red[tid], red[tid + s]); __syncthreads(); }
  mx = red[0]; __syncthreads();
  float sm = 0.f;
  for (int t = tid; t < T; t += 128) {
    float v = base[(size_t)t * Dtot];
    if (mask) v += (1.f - mask[b * T + t]) * (-1000000.0f);
    sm += __expf(v - mx);
  }
  red[tid] = sm; __syncthreads();
  for (int s = 64; s > 0; s >>= 1) { if (tid < s) red[tid] += red[tid + s]; __syncthreads(); }
  float inv = 1.f / red[0]; __syncthreads();
  for (int t = tid; t < T; t += 128) {
    float v = base[(size_t)t * Dtot];
    if (mask) v += (1.f - mask[b * T + t]) * (-1000000.0f);
    base[(size_t)t * Dtot] = __expf(v - mx) * inv;
  }
}

// v *= mask[b,t]
__global__ void vmask_k(float* __restrict__ V, const float* __restrict__ mask, int n) {
  int idx = blockIdx.x * blockDim.x + threadIdx.x;
  if (idx < n) V[idx] *= mask[idx >> 9];       // D = 512
}

// fp32 -> bf16 elementwise
__global__ void cvt_bf16_k(const float* __restrict__ X, bf16_t* __restrict__ O, int n) {
  int idx = blockIdx.x * blockDim.x + threadIdx.x;
  if (idx < n) O[idx] = f2bf(X[idx]);
}

// fp32 [K,N] -> bf16 transposed [N,K]
__global__ void cvtT_k(const float* __restrict__ W, bf16_t* __restrict__ Wt,
                       int K, int N) {
  int idx = blockIdx.x * blockDim.x + threadIdx.x;
  if (idx < K * N) {
    int k = idx / N, n = idx % N;
    Wt[(size_t)n * K + k] = f2bf(W[idx]);
  }
}

// timestep embedding
__global__ __launch_bounds__(256) void tstep_k(const int* __restrict__ ts, float* __restrict__ te) {
  int b = blockIdx.x, i = threadIdx.x;
  float t = (float)ts[b];
  float freq = __expf(-9.210340371976184f * (float)i / 256.f);   // ln(1e4)
  float a = t * freq;
  te[(size_t)b * Dd + i]       = cosf(a);
  te[(size_t)b * Dd + 256 + i] = sinf(a);
}

// small scalar GEMM: O[m,n] = bias[n] + sum_k act(A[m,k]) * W[k,n]
__global__ void small_gemm_k(const float* __restrict__ A, const float* __restrict__ W,
                             const float* __restrict__ bias, float* __restrict__ O,
                             int M, int K, int N, int siluA) {
  int idx = blockIdx.x * blockDim.x + threadIdx.x;
  if (idx >= M * N) return;
  int m = idx / N, n = idx % N;
  float s = bias[n];
  for (int k = 0; k < K; k++) {
    float a = A[(size_t)m * K + k];
    if (siluA) a = a / (1.f + __expf(-a));
    s += a * W[(size_t)k * N + n];
  }
  O[idx] = s;
}

// ---------------------------------------------------------------------------
extern "C" void kernel_launch(void* const* d_in, const int* in_sizes, int n_in,
                              void* d_out, int out_size, void* d_ws, size_t ws_size,
                              hipStream_t stream) {
  (void)in_sizes; (void)n_in; (void)out_size; (void)ws_size;

  const float* x_in   = (const float*)d_in[0];   // [B,T,D]
  const float* xf     = (const float*)d_in[1];   // [B,NTXT,LTXT]
  const int*   tsteps = (const int*)d_in[2];     // [B]
  const float* mask   = (const float*)d_in[3];   // [B,T,1]

  // params: jax pytree flatten order (sorted dict keys at each level)
  const float* ca_kb    = (const float*)d_in[4];
  const float* ca_kw    = (const float*)d_in[5];
  const float* ca_nb    = (const float*)d_in[6];
  const float* ca_ng    = (const float*)d_in[7];
  const float* ca_qb    = (const float*)d_in[8];
  const float* ca_qw    = (const float*)d_in[9];
  const float* ca_so_eb = (const float*)d_in[10];
  const float* ca_so_ew = (const float*)d_in[11];
  const float* ca_so_nb = (const float*)d_in[12];
  const float* ca_so_ng = (const float*)d_in[13];
  const float* ca_so_ob = (const float*)d_in[14];
  const float* ca_so_ow = (const float*)d_in[15];
  const float* ca_tnb   = (const float*)d_in[16];
  const float* ca_tng   = (const float*)d_in[17];
  const float* ca_vb    = (const float*)d_in[18];
  const float* ca_vw    = (const float*)d_in[19];
  const float* f_b1     = (const float*)d_in[20];
  const float* f_b2     = (const float*)d_in[21];
  const float* f_so_eb  = (const float*)d_in[22];
  const float* f_so_ew  = (const float*)d_in[23];
  const float* f_so_nb  = (const float*)d_in[24];
  const float* f_so_ng  = (const float*)d_in[25];
  const float* f_so_ob  = (const float*)d_in[26];
  const float* f_so_ow  = (const float*)d_in[27];
  const float* f_w1     = (const float*)d_in[28];
  const float* f_w2     = (const float*)d_in[29];
  const float* sa_kb    = (const float*)d_in[30];
  const float* sa_kw    = (const float*)d_in[31];
  const float* sa_nb    = (const float*)d_in[32];
  const float* sa_ng    = (const float*)d_in[33];
  const float* sa_qb    = (const float*)d_in[34];
  const float* sa_qw    = (const float*)d_in[35];
  const float* sa_so_eb = (const float*)d_in[36];
  const float* sa_so_ew = (const float*)d_in[37];
  const float* sa_so_nb = (const float*)d_in[38];
  const float* sa_so_ng = (const float*)d_in[39];
  const float* sa_so_ob = (const float*)d_in[40];
  const float* sa_so_ow = (const float*)d_in[41];
  const float* sa_vb    = (const float*)d_in[42];
  const float* sa_vw    = (const float*)d_in[43];
  const float* t_b1     = (const float*)d_in[44];
  const float* t_b2     = (const float*)d_in[45];
  const float* t_w1     = (const float*)d_in[46];
  const float* t_w2     = (const float*)d_in[47];

  // ----- workspace carve-up -----
  char* wsp = (char*)d_ws;
  auto alloc = [&](size_t bytes) -> char* {
    char* p = wsp; wsp += (bytes + 255) & ~(size_t)255; return p;
  };
  bf16_t* xnbf  = (bf16_t*)alloc((size_t)Mx * Dd * 2);
  bf16_t* midbf = (bf16_t*)alloc((size_t)Mx * FFNd * 2);
  float*  qf    = (float*)alloc((size_t)Mx * Dd * 4);
  float*  kf    = (float*)alloc((size_t)Mx * Dd * 4);
  float*  vf    = (float*)alloc((size_t)Mx * Dd * 4);
  float*  yf    = (float*)alloc((size_t)Mx * Dd * 4);
  float*  kv    = (float*)alloc((size_t)Bb * Hh * DH * DH * 4);
  float*  te    = (float*)alloc((size_t)Bb * Dd * 4);
  float*  t1    = (float*)alloc((size_t)Bb * TEd * 4);
  float*  emb   = (float*)alloc((size_t)Bb * TEd * 4);
  float*  eSS   = (float*)alloc((size_t)Bb * 2 * Dd * 4);
  // transposed bf16 weights (per layer, reused)
  bf16_t* wqT   = (bf16_t*)alloc((size_t)Dd * Dd * 2);
  bf16_t* wkT   = (bf16_t*)alloc((size_t)Dd * Dd * 2);
  bf16_t* wvT   = (bf16_t*)alloc((size_t)Dd * Dd * 2);
  bf16_t* wowT  = (bf16_t*)alloc((size_t)Dd * Dd * 2);
  bf16_t* w1T   = (bf16_t*)alloc((size_t)Dd * FFNd * 2);
  bf16_t* w2T   = (bf16_t*)alloc((size_t)Dd * FFNd * 2);
  // xfn bf16 with 128 rows slack (TDM tiles may read past M=2464)
  bf16_t* xfnbf = (bf16_t*)alloc((size_t)(Bb * NTXT + 128) * LTXT * 2);

  float* x = (float*)d_out;   // running activation, updated in place

  hipMemcpyAsync(x, x_in, (size_t)Mx * Dd * sizeof(float),
                 hipMemcpyDeviceToDevice, stream);

  // ----- time embedding MLP -----
  tstep_k<<<Bb, 256, 0, stream>>>(tsteps, te);
  small_gemm_k<<<(Bb * TEd + 255) / 256, 256, 0, stream>>>(te, t_w1, t_b1, t1, Bb, Dd, TEd, 0);
  small_gemm_k<<<(Bb * TEd + 255) / 256, 256, 0, stream>>>(t1, t_w2, t_b2, emb, Bb, TEd, TEd, 1);

  const dim3 g512(Dd / 64, (Mx + 127) / 128);        // [16384 x 512]
  const dim3 gffn(FFNd / 64, (Mx + 127) / 128);      // [16384 x 2048]
  const dim3 gkv(Dd / 64, (Bb * NTXT + 127) / 128);  // [2464 x 512]
  const dim3 gqatt(Bb * Hh, (Tt + 127) / 128);
  const int gT512 = (Dd * Dd + 255) / 256;           // 512x512 transpose grid
  const int gTffn = (Dd * FFNd + 255) / 256;
  const int gTkv  = (LTXT * Dd + 255) / 256;

  for (int l = 0; l < NL; l++) {
    // ================= self attention =================
    cvtT_k<<<gT512, 256, 0, stream>>>(sa_qw + (size_t)l * Dd * Dd, wqT, Dd, Dd);
    cvtT_k<<<gT512, 256, 0, stream>>>(sa_kw + (size_t)l * Dd * Dd, wkT, Dd, Dd);
    cvtT_k<<<gT512, 256, 0, stream>>>(sa_vw + (size_t)l * Dd * Dd, wvT, Dd, Dd);
    cvtT_k<<<gT512, 256, 0, stream>>>(sa_so_ow + (size_t)l * Dd * Dd, wowT, Dd, Dd);
    layernorm_bf16_k<<<Mx, 256, 0, stream>>>(x, sa_ng + l * Dd, sa_nb + l * Dd, xnbf, Dd);
    gemm_tdm_k<0, float><<<g512, 256, 0, stream>>>(xnbf, wqT, sa_qb + l * Dd, nullptr, qf, Mx, Dd, Dd);
    gemm_tdm_k<0, float><<<g512, 256, 0, stream>>>(xnbf, wkT, sa_kb + l * Dd, nullptr, kf, Mx, Dd, Dd);
    gemm_tdm_k<0, float><<<g512, 256, 0, stream>>>(xnbf, wvT, sa_vb + l * Dd, nullptr, vf, Mx, Dd, Dd);
    softmax_dh_k<<<Mx * Hh, 64, 0, stream>>>(qf);
    softmax_tok_k<<<Bb * Dd, 128, 0, stream>>>(kf, mask, Tt, Dd);
    vmask_k<<<(Mx * Dd + 255) / 256, 256, 0, stream>>>(vf, mask, Mx * Dd);
    attn_kv_k<<<Bb * Hh, 128, 0, stream>>>(kf, vf, kv, Tt, Dd);
    attn_qatt_k<<<gqatt, 256, 0, stream>>>(qf, kv, yf, Tt, Dd);
    small_gemm_k<<<(Bb * 2 * Dd + 255) / 256, 256, 0, stream>>>(
        emb, sa_so_ew + (size_t)l * TEd * 2 * Dd, sa_so_eb + l * 2 * Dd, eSS, Bb, TEd, 2 * Dd, 1);
    mod_silu_k<<<Mx, 256, 0, stream>>>(yf, sa_so_ng + l * Dd, sa_so_nb + l * Dd, eSS, xnbf);
    gemm_tdm_k<2, float><<<g512, 256, 0, stream>>>(xnbf, wowT, sa_so_ob + l * Dd, x, x, Mx, Dd, Dd);

    // ================= cross attention =================
    cvtT_k<<<gT512, 256, 0, stream>>>(ca_qw + (size_t)l * Dd * Dd, wqT, Dd, Dd);
    cvtT_k<<<gTkv, 256, 0, stream>>>(ca_kw + (size_t)l * LTXT * Dd, wkT, LTXT, Dd);
    cvtT_k<<<gTkv, 256, 0, stream>>>(ca_vw + (size_t)l * LTXT * Dd, wvT, LTXT, Dd);
    cvtT_k<<<gT512, 256, 0, stream>>>(ca_so_ow + (size_t)l * Dd * Dd, wowT, Dd, Dd);
    layernorm_bf16_k<<<Mx, 256, 0, stream>>>(x, ca_ng + l * Dd, ca_nb + l * Dd, xnbf, Dd);
    gemm_tdm_k<0, float><<<g512, 256, 0, stream>>>(xnbf, wqT, ca_qb + l * Dd, nullptr, qf, Mx, Dd, Dd);
    layernorm_bf16_k<<<Bb * NTXT, 256, 0, stream>>>(
        xf, ca_tng + l * LTXT, ca_tnb + l * LTXT, xfnbf, LTXT);
    gemm_tdm_k<0, float><<<gkv, 256, 0, stream>>>(xfnbf, wkT, ca_kb + l * Dd, nullptr, kf, Bb * NTXT, Dd, LTXT);
    gemm_tdm_k<0, float><<<gkv, 256, 0, stream>>>(xfnbf, wvT, ca_vb + l * Dd, nullptr, vf, Bb * NTXT, Dd, LTXT);
    softmax_dh_k<<<Mx * Hh, 64, 0, stream>>>(qf);
    softmax_tok_k<<<Bb * Dd, 128, 0, stream>>>(kf, nullptr, NTXT, Dd);
    attn_kv_k<<<Bb * Hh, 128, 0, stream>>>(kf, vf, kv, NTXT, Dd);
    attn_qatt_k<<<gqatt, 256, 0, stream>>>(qf, kv, yf, Tt, Dd);
    small_gemm_k<<<(Bb * 2 * Dd + 255) / 256, 256, 0, stream>>>(
        emb, ca_so_ew + (size_t)l * TEd * 2 * Dd, ca_so_eb + l * 2 * Dd, eSS, Bb, TEd, 2 * Dd, 1);
    mod_silu_k<<<Mx, 256, 0, stream>>>(yf, ca_so_ng + l * Dd, ca_so_nb + l * Dd, eSS, xnbf);
    gemm_tdm_k<2, float><<<g512, 256, 0, stream>>>(xnbf, wowT, ca_so_ob + l * Dd, x, x, Mx, Dd, Dd);

    // ================= FFN =================
    cvtT_k<<<gTffn, 256, 0, stream>>>(f_w1 + (size_t)l * Dd * FFNd, w1T, Dd, FFNd);
    cvtT_k<<<gTffn, 256, 0, stream>>>(f_w2 + (size_t)l * FFNd * Dd, w2T, FFNd, Dd);
    cvtT_k<<<gT512, 256, 0, stream>>>(f_so_ow + (size_t)l * Dd * Dd, wowT, Dd, Dd);
    cvt_bf16_k<<<(Mx * Dd + 255) / 256, 256, 0, stream>>>(x, xnbf, Mx * Dd);
    gemm_tdm_k<1, bf16_t><<<gffn, 256, 0, stream>>>(   // fused exact GELU -> bf16
        xnbf, w1T, f_b1 + l * FFNd, nullptr, midbf, Mx, FFNd, Dd);
    gemm_tdm_k<0, float><<<g512, 256, 0, stream>>>(midbf, w2T, f_b2 + l * Dd, nullptr, yf, Mx, Dd, FFNd);
    small_gemm_k<<<(Bb * 2 * Dd + 255) / 256, 256, 0, stream>>>(
        emb, f_so_ew + (size_t)l * TEd * 2 * Dd, f_so_eb + l * 2 * Dd, eSS, Bb, TEd, 2 * Dd, 1);
    mod_silu_k<<<Mx, 256, 0, stream>>>(yf, f_so_ng + l * Dd, f_so_nb + l * Dd, eSS, xnbf);
    gemm_tdm_k<2, float><<<g512, 256, 0, stream>>>(xnbf, wowT, f_so_ob + l * Dd, x, x, Mx, Dd, Dd);
  }
}
